// CausalSelfAttention_6150393168515
// MI455X (gfx1250) — compile-verified, matrix operand order checked by reading, not run on hardware
//
#include <hip/hip_runtime.h>

// ---------------------------------------------------------------------------
// Problem constants (match reference)
// ---------------------------------------------------------------------------
#define B_SZ   4
#define T_SZ   1024
#define C_SZ   1024
#define NHEAD  16
#define HD     64          // C / NHEAD
#define INTD   64

typedef __attribute__((ext_vector_type(16))) __bf16 v16bf;
typedef __attribute__((ext_vector_type(8)))  float  v8f;

__device__ __forceinline__ v8f wmma_bf16(v16bf a, v16bf b, v8f c) {
  // D = A(16x32 bf16) * B(32x16 bf16) + C(16x16 f32)
  return __builtin_amdgcn_wmma_f32_16x16x32_bf16(
      /*neg_a=*/false, a, /*neg_b=*/false, b,
      /*c_mod=*/(short)0, c, /*reuse_a=*/false, /*reuse_b=*/false);
}

// Async copy 16 bytes global -> LDS (CDNA5 GLOBAL_LOAD_ASYNC_TO_LDS_B128,
// tracked by ASYNCcnt). VDST VGPR carries the LDS byte address.
__device__ __forceinline__ void async_cp16(const void* gsrc, void* lds_dst) {
  const unsigned lds =
      (unsigned)(reinterpret_cast<unsigned long long>(lds_dst) & 0xffffffffull);
  asm volatile("global_load_async_to_lds_b128 %0, %1, off"
               :
               : "v"(lds), "v"(gsrc)
               : "memory");
}

// Build a bf16 fragment from 16 consecutive f32 values (flash P-fragment).
__device__ __forceinline__ v16bf frag_from_f32(const float* p) {
  v16bf f;
#pragma unroll
  for (int j = 0; j < 16; ++j) f[j] = (__bf16)p[j];
  return f;
}

// ---------------------------------------------------------------------------
// Elementwise f32 -> bf16 converter (n % 4 == 0 for all uses here)
// ---------------------------------------------------------------------------
__global__ __launch_bounds__(256)
void cvt_f32_to_bf16(const float* __restrict__ in, unsigned short* __restrict__ out,
                     long n) {
  __bf16* o = reinterpret_cast<__bf16*>(out);
  const long i = ((long)blockIdx.x * blockDim.x + threadIdx.x) * 4;
  const long stride = (long)gridDim.x * blockDim.x * 4;
  for (long e = i; e < n; e += stride) {
    const float4 v = *reinterpret_cast<const float4*>(&in[e]);
    o[e + 0] = (__bf16)v.x;
    o[e + 1] = (__bf16)v.y;
    o[e + 2] = (__bf16)v.z;
    o[e + 3] = (__bf16)v.w;
  }
}

// ---------------------------------------------------------------------------
// Tiled GEMM on pre-converted bf16:  C(MxN) = alpha * A(MxK) * B(NxK)^T + addv
// Block: 128 threads (4 waves). Block tile 64x64, K-step 64.
// Double-buffered async global->LDS staging overlapped with WMMA; output
// written as f32 (Cf) or bf16 (Cb). Assumes M,N % 64 == 0, K % 64 == 0.
// ---------------------------------------------------------------------------
__global__ __launch_bounds__(128)
void gemm_bf16_wmma(const unsigned short* __restrict__ Au,
                    const unsigned short* __restrict__ Bu,
                    float* __restrict__ Cf, unsigned short* __restrict__ Cb,
                    int N, int K,
                    long sAb, long sBb, long sCb,
                    float alpha, const float* __restrict__ add_scalar) {
  __shared__ alignas(32) __bf16 As[2][64 * 64];
  __shared__ alignas(32) __bf16 Bs[2][64 * 64];
  const __bf16* A  = reinterpret_cast<const __bf16*>(Au) + (long)blockIdx.z * sAb;
  const __bf16* Bw = reinterpret_cast<const __bf16*>(Bu) + (long)blockIdx.z * sBb;
  if (Cf) Cf += (long)blockIdx.z * sCb;
  __bf16* Cbb = Cb ? reinterpret_cast<__bf16*>(Cb) + (long)blockIdx.z * sCb : nullptr;
  const int n0   = blockIdx.x * 64;
  const int m0   = blockIdx.y * 64;
  const int tid  = threadIdx.x;
  const int wave = tid >> 5;
  const int lane = tid & 31;
  const int lrow = lane & 15;
  const int lko  = (lane >> 4) * 16;
  const float addv = add_scalar ? add_scalar[0] : 0.0f;

  // each thread stages 32 consecutive bf16 (4 async b128) per matrix per tile
  const int srow = (tid * 32) >> 6;
  const int scol = (tid * 32) & 63;
  auto issue_tile = [&](int pb, int kk) {
    const __bf16* ag = &A [(long)(m0 + srow) * K + kk + scol];
    const __bf16* bg = &Bw[(long)(n0 + srow) * K + kk + scol];
    __bf16* ad = &As[pb][srow * 64 + scol];
    __bf16* bd = &Bs[pb][srow * 64 + scol];
#pragma unroll
    for (int c = 0; c < 4; ++c) {
      async_cp16(ag + c * 8, ad + c * 8);
      async_cp16(bg + c * 8, bd + c * 8);
    }
  };

  v8f acc[4] = {v8f{}, v8f{}, v8f{}, v8f{}};

  issue_tile(0, 0);
  int pb = 0;
  for (int kk = 0; kk < K; kk += 64, pb ^= 1) {
    const bool more = (kk + 64) < K;
    if (more) {
      issue_tile(pb ^ 1, kk + 64);                    // prefetch next tile
      asm volatile("s_wait_asynccnt 8" ::: "memory"); // in-order: tile pb done
    } else {
      asm volatile("s_wait_asynccnt 0" ::: "memory");
    }
    __syncthreads();

    const v16bf a0 = *reinterpret_cast<const v16bf*>(&As[pb][(wave * 16 + lrow) * 64 + lko]);
    const v16bf a1 = *reinterpret_cast<const v16bf*>(&As[pb][(wave * 16 + lrow) * 64 + 32 + lko]);
#pragma unroll
    for (int nb = 0; nb < 4; ++nb) {
      const v16bf b0 = *reinterpret_cast<const v16bf*>(&Bs[pb][(nb * 16 + lrow) * 64 + lko]);
      const v16bf b1 = *reinterpret_cast<const v16bf*>(&Bs[pb][(nb * 16 + lrow) * 64 + 32 + lko]);
      acc[nb] = wmma_bf16(a0, b0, acc[nb]);
      acc[nb] = wmma_bf16(a1, b1, acc[nb]);
    }
    __syncthreads();  // all waves done reading buf pb before it is re-staged
  }

  // C/D layout: VGPR r -> row r (lanes 0-15) or row r+8 (lanes 16-31)
  const int ro = (lane >> 4) * 8;
#pragma unroll
  for (int nb = 0; nb < 4; ++nb) {
#pragma unroll
    for (int r = 0; r < 8; ++r) {
      const int row = m0 + wave * 16 + r + ro;
      const int col = n0 + nb * 16 + lrow;
      const float v = alpha * acc[nb][r] + addv;
      if (Cf) Cf[(long)row * N + col] = v;
      else    Cbb[(long)row * N + col] = (__bf16)v;
    }
  }
}

// ---------------------------------------------------------------------------
// Extract V from bf16 qkv and transpose per (b,h): (T, hd) -> (hd, T)
// ---------------------------------------------------------------------------
__global__ __launch_bounds__(256)
void extract_vT(const unsigned short* __restrict__ qkvb_u,
                unsigned short* __restrict__ vT_u) {
  __shared__ __bf16 tile[32][33];
  const __bf16* q = reinterpret_cast<const __bf16*>(qkvb_u);
  __bf16* vT = reinterpret_cast<__bf16*>(vT_u);
  const int t0 = blockIdx.x * 32;
  const int d0 = blockIdx.y * 32;
  const int bh = blockIdx.z;
  const int b = bh / NHEAD, h = bh % NHEAD;
  const int tx = threadIdx.x, ty = threadIdx.y;      // (32, 8)
  const long base = (long)b * T_SZ * 3 * C_SZ + 2 * C_SZ + h * HD;
#pragma unroll
  for (int k = 0; k < 32; k += 8)
    tile[ty + k][tx] = q[base + (long)(t0 + ty + k) * 3 * C_SZ + d0 + tx];
  __syncthreads();
  const long obase = (long)bh * HD * T_SZ;
#pragma unroll
  for (int k = 0; k < 32; k += 8)
    vT[obase + (long)(d0 + ty + k) * T_SZ + t0 + tx] = tile[tx][ty + k];
}

// ---------------------------------------------------------------------------
// Fused 2-way entmax-bisect + causal fill + cumprod + log mask.
// One thread per (b, s) column; sequential over the query dim t.
// ---------------------------------------------------------------------------
__global__ __launch_bounds__(256)
void entmax_cumprod(const float* __restrict__ scores,
                    float* __restrict__ cumout,
                    float* __restrict__ maskout) {
  const int idx = blockIdx.x * blockDim.x + threadIdx.x;
  if (idx >= B_SZ * T_SZ) return;
  const int b = idx / T_SZ;
  const int s = idx % T_SZ;
  const float am1  = 1.0000000000287557e-06f;  // ALPHA - 1
  const float powr = 9.9999999997124500e+05f;  // 1 / (ALPHA - 1)
  const float* sp = &scores[(long)b * T_SZ * T_SZ + s];
  float* cp = &cumout [(long)b * T_SZ * T_SZ + s];
  float* mp = &maskout[(long)b * T_SZ * T_SZ + s];
  float cum = 1.0f;
  for (int t = 0; t < T_SZ; ++t) {
    float p;
    if (s < t) {
      const float Xa0   = sp[(long)t * T_SZ] * am1;  // Xa = [x*(a-1), 0]
      const float mx    = fmaxf(Xa0, 0.0f);
      float tau_lo      = mx - 1.0f;
      const float tauhi = mx - __powf(0.5f, powr);   // (1/d)^power, d=2
      float dm          = tauhi - tau_lo;
      float p0 = 0.0f, p1 = 0.0f;
#pragma unroll 1
      for (int i = 0; i < 50; ++i) {
        dm *= 0.5f;
        const float tau = tau_lo + dm;
        p0 = __powf(fmaxf(Xa0 - tau, 0.0f), powr);
        p1 = __powf(fmaxf(0.0f - tau, 0.0f), powr);
        if (p0 + p1 - 1.0f >= 0.0f) tau_lo = tau;
      }
      const float den = p0 + p1;
      p = (den > 0.0f) ? (p0 / den) : (Xa0 >= 0.0f ? 1.0f : 0.0f);
    } else {
      p = 1.0f;
    }
    cum *= p;
    cp[(long)t * T_SZ] = cum;
    mp[(long)t * T_SZ] = (s <= t) ? __logf(cum + 1e-40f) : -__builtin_inff();
  }
}

// ---------------------------------------------------------------------------
// Flash attention: 4 waves/block, each wave owns a 16-query tile; the block
// shares one async-staged K tile (32x64) and V^T tile (64x32) per key step.
// S = Q·K^T (WMMA), vals = S/8 + attn_mask, online softmax (per-wave LDS
// scratch), O += P·V (WMMA with contiguous V^T fragments). bf16 output.
// ---------------------------------------------------------------------------
__global__ __launch_bounds__(128)
void flash_attn(const unsigned short* __restrict__ qkvb_u,  // bf16 (B,T,3C)
                const unsigned short* __restrict__ vT_u,    // bf16 (B,H,HD,T)
                const float* __restrict__ mask,             // f32 (B,T,T)
                unsigned short* __restrict__ ypb_u) {       // bf16 (B,T,C)
  __shared__ alignas(32) __bf16 Kt[32 * HD];       // keys x dims
  __shared__ alignas(32) __bf16 VtT[HD * 32];      // dims x keys
  __shared__ float Sm[4][16 * 32];                 // per-wave scratch
  __shared__ float scale_l[4][16];
  __shared__ float lsum_l[4][16];

  const __bf16* qkvb = reinterpret_cast<const __bf16*>(qkvb_u);
  const __bf16* vT   = reinterpret_cast<const __bf16*>(vT_u);
  __bf16* ypb        = reinterpret_cast<__bf16*>(ypb_u);

  const int h = blockIdx.y, b = blockIdx.z;
  const int bq0  = blockIdx.x * 64;
  const int tid  = threadIdx.x;
  const int wave = tid >> 5;
  const int lane = tid & 31;
  const int q0   = bq0 + wave * 16;
  const int lrow = lane & 15;
  const int lko  = (lane >> 4) * 16;
  const long base  = (long)b * T_SZ * 3 * C_SZ;
  const long vbase = (long)(b * NHEAD + h) * HD * T_SZ;

  // Q fragments, direct bf16 global loads (1/sqrt(hd) folded into softmax)
  v16bf qf[2];
#pragma unroll
  for (int f = 0; f < 2; ++f)
    qf[f] = *reinterpret_cast<const v16bf*>(
        &qkvb[base + (long)(q0 + lrow) * 3 * C_SZ + h * HD + f * 32 + lko]);

  v8f accO[4] = {v8f{}, v8f{}, v8f{}, v8f{}};
  float m_i = -3.0e38f, l_i = 0.0f;
  const float* maskb = &mask[(long)b * T_SZ * T_SZ];
  const int ktiles = (bq0 + 63) / 32 + 1;

  for (int kt = 0; kt < ktiles; ++kt) {
    const int k0 = kt * 32;
    // shared async staging: K tile (2048 bf16) + V^T tile (2048 bf16)
    {
      const int e = tid * 16;
      {
        const int row = e >> 6, col = e & 63;
        const __bf16* g = &qkvb[base + (long)(k0 + row) * 3 * C_SZ + C_SZ + h * HD + col];
        __bf16* d = &Kt[row * 64 + col];
        async_cp16(g, d); async_cp16(g + 8, d + 8);
      }
      {
        const int row = e >> 5, col = e & 31;
        const __bf16* g = &vT[vbase + (long)row * T_SZ + k0 + col];
        __bf16* d = &VtT[row * 32 + col];
        async_cp16(g, d); async_cp16(g + 8, d + 8);
      }
    }
    asm volatile("s_wait_asynccnt 0" ::: "memory");
    __syncthreads();

    if (k0 <= q0 + 15) {  // wave-uniform causal predicate
      // S tiles (16 queries x 32 keys)
#pragma unroll
      for (int j = 0; j < 2; ++j) {
        v8f s = {};
#pragma unroll
        for (int kc = 0; kc < 2; ++kc) {
          const v16bf bf = *reinterpret_cast<const v16bf*>(
              &Kt[(j * 16 + lrow) * 64 + kc * 32 + lko]);
          s = wmma_bf16(qf[kc], bf, s);
        }
#pragma unroll
        for (int r = 0; r < 8; ++r)
          Sm[wave][(r + ((lane >> 4) * 8)) * 32 + j * 16 + lrow] = s[r];
      }

      // online softmax: lanes 0..15 each own one query row (wave-internal DS)
      if (lane < 16) {
        const int t = q0 + lane;
        float rowm = m_i;
        float vals[32];
#pragma unroll
        for (int c = 0; c < 32; ++c) {
          const float v = Sm[wave][lane * 32 + c] * 0.125f +
                          maskb[(long)t * T_SZ + (k0 + c)];
          vals[c] = v;
          rowm = fmaxf(rowm, v);
        }
        const float sc = __expf(m_i - rowm);
        float rs = 0.0f;
#pragma unroll
        for (int c = 0; c < 32; ++c) {
          const float pv = __expf(vals[c] - rowm);
          Sm[wave][lane * 32 + c] = pv;  // Sm now holds P
          rs += pv;
        }
        l_i = l_i * sc + rs;
        m_i = rowm;
        scale_l[wave][lane] = sc;
      }

      // rescale O accumulators
      {
        const int ro = (lane >> 4) * 8;
#pragma unroll
        for (int r = 0; r < 8; ++r) {
          const float sc = scale_l[wave][r + ro];
#pragma unroll
          for (int nb = 0; nb < 4; ++nb) accO[nb][r] *= sc;
        }
      }

      // P fragment (f32->bf16) and contiguous V^T fragments; O += P·V
      const v16bf pf = frag_from_f32(&Sm[wave][lrow * 32 + lko]);
#pragma unroll
      for (int nb = 0; nb < 4; ++nb) {
        const v16bf vf = *reinterpret_cast<const v16bf*>(
            &VtT[(nb * 16 + lrow) * 32 + lko]);
        accO[nb] = wmma_bf16(pf, vf, accO[nb]);
      }
    }
    __syncthreads();  // protect shared K/V tiles before re-staging
  }

  if (lane < 16) lsum_l[wave][lane] = l_i;
  __syncthreads();

  const int ro = (lane >> 4) * 8;
#pragma unroll
  for (int r = 0; r < 8; ++r) {
    const float inv = 1.0f / lsum_l[wave][r + ro];
    const int row = q0 + r + ro;
#pragma unroll
    for (int nb = 0; nb < 4; ++nb)
      ypb[(long)b * T_SZ * C_SZ + (long)row * C_SZ + h * HD + nb * 16 + lrow] =
          (__bf16)(accO[nb][r] * inv);
  }
}

// ---------------------------------------------------------------------------
// Host-side orchestration
// ---------------------------------------------------------------------------
static inline dim3 cvt_grid(long n) {
  long blocks = (n / 4 + 255) / 256;
  if (blocks > 65535) blocks = 65535;
  return dim3((unsigned)blocks);
}

extern "C" void kernel_launch(void* const* d_in, const int* in_sizes, int n_in,
                              void* d_out, int out_size, void* d_ws, size_t ws_size,
                              hipStream_t stream) {
  (void)in_sizes; (void)n_in; (void)out_size; (void)ws_size;
  const float* x        = (const float*)d_in[0];   // (B,T,C)
  const float* W_attn   = (const float*)d_in[1];   // (3C,C)
  const float* W_proj   = (const float*)d_in[2];   // (C,C)
  const float* Wq_int   = (const float*)d_in[3];   // (INTD,C)
  const float* Wk_int   = (const float*)d_in[4];   // (INTD,C)
  const float* int_bias = (const float*)d_in[5];   // (1,)

  const long BT = (long)B_SZ * T_SZ;           // 4096
  const long yN = BT * C_SZ;
  const long mN = (long)B_SZ * T_SZ * T_SZ;

  float* y        = (float*)d_out;
  float* cumprobs = y + yN;
  float* attn_msk = cumprobs + mN;

  // f32 workspace
  float* wsf    = (float*)d_ws;
  float* scores = wsf;  wsf += mN;
  // bf16 workspace (stored as u16)
  unsigned short* p    = (unsigned short*)wsf;
  unsigned short* xb   = p;  p += BT * C_SZ;
  unsigned short* Wab  = p;  p += 3L * C_SZ * C_SZ;
  unsigned short* Wpb  = p;  p += (long)C_SZ * C_SZ;
  unsigned short* Wqb  = p;  p += (long)INTD * C_SZ;
  unsigned short* Wkb  = p;  p += (long)INTD * C_SZ;
  unsigned short* qkvb = p;  p += BT * 3 * C_SZ;
  unsigned short* vTb  = p;  p += yN;              // (B,H,HD,T)
  unsigned short* qib  = p;  p += BT * INTD;
  unsigned short* kib  = p;  p += BT * INTD;
  unsigned short* ypb  = p;

  // 0) one-time bf16 conversions of GEMM operands
  cvt_f32_to_bf16<<<cvt_grid(yN), 256, 0, stream>>>(x, xb, yN);
  cvt_f32_to_bf16<<<cvt_grid(3L * C_SZ * C_SZ), 256, 0, stream>>>(W_attn, Wab, 3L * C_SZ * C_SZ);
  cvt_f32_to_bf16<<<cvt_grid((long)C_SZ * C_SZ), 256, 0, stream>>>(W_proj, Wpb, (long)C_SZ * C_SZ);
  cvt_f32_to_bf16<<<cvt_grid((long)INTD * C_SZ), 256, 0, stream>>>(Wq_int, Wqb, (long)INTD * C_SZ);
  cvt_f32_to_bf16<<<cvt_grid((long)INTD * C_SZ), 256, 0, stream>>>(Wk_int, Wkb, (long)INTD * C_SZ);

  // 1) qkv = x @ W_attn^T  (bf16 out, 4096 x 3072 x 1024)
  gemm_bf16_wmma<<<dim3(3 * C_SZ / 64, BT / 64, 1), 128, 0, stream>>>(
      xb, Wab, nullptr, qkvb, 3 * C_SZ, C_SZ, 0, 0, 0, 1.0f, nullptr);

  // 1b) V^T extraction per (b,h): (T,hd) -> (hd,T)
  extract_vT<<<dim3(T_SZ / 32, HD / 32, B_SZ * NHEAD), dim3(32, 8), 0, stream>>>(
      qkvb, vTb);

  // 2) q_int / k_int = x @ W*_int^T  (bf16 out, 4096 x 64 x 1024)
  gemm_bf16_wmma<<<dim3(1, BT / 64, 1), 128, 0, stream>>>(
      xb, Wqb, nullptr, qib, INTD, C_SZ, 0, 0, 0, 1.0f, nullptr);
  gemm_bf16_wmma<<<dim3(1, BT / 64, 1), 128, 0, stream>>>(
      xb, Wkb, nullptr, kib, INTD, C_SZ, 0, 0, 0, 1.0f, nullptr);

  // 3) scores[b] = q_int[b] @ k_int[b]^T / 8 + bias  (f32 out, batched)
  gemm_bf16_wmma<<<dim3(T_SZ / 64, T_SZ / 64, B_SZ), 128, 0, stream>>>(
      qib, kib, scores, nullptr, T_SZ, INTD,
      (long)T_SZ * INTD, (long)T_SZ * INTD, (long)T_SZ * T_SZ,
      0.125f, int_bias);

  // 4) entmax bisect + cumprod + log-mask -> cumprobs, attn_mask
  entmax_cumprod<<<dim3((B_SZ * T_SZ + 255) / 256), 256, 0, stream>>>(
      scores, cumprobs, attn_msk);

  // 5) masked softmax attention -> ypre (bf16)
  flash_attn<<<dim3(T_SZ / 64, NHEAD, B_SZ), 128, 0, stream>>>(
      qkvb, vTb, attn_msk, ypb);

  // 6) y = ypre @ W_proj^T  (f32 out, 4096 x 1024 x 1024)
  gemm_bf16_wmma<<<dim3(C_SZ / 64, BT / 64, 1), 128, 0, stream>>>(
      ypb, Wpb, y, nullptr, C_SZ, C_SZ, 0, 0, 0, 1.0f, nullptr);
}
